// GIN_52140902974155
// MI455X (gfx1250) — compile-verified, matrix operand order checked by reading, not run on hardware
//
#include <hip/hip_runtime.h>
#include <hip/hip_bf16.h>

typedef __attribute__((ext_vector_type(16))) _Float16 v16h;
typedef __attribute__((ext_vector_type(4)))  _Float16 v4h;
typedef __attribute__((ext_vector_type(8)))  float    v8f;

#define HDIM    128
#define NNODES  50000   // multiple of 16 -> per-tile validity is uniform
#define NEDGES  800000  // multiple of 64
#define NLAYERS 3

// ---------------------------------------------------------------------------
// Fragment load. LDS tile is row-major [rows][K] f16 (K even).
// A-fragment: lane holds row M = lane&15; VGPR v holds K-pair
//   k0 = kk + (v&3)*2 + (lane&16 ? 8 : 0) + (v>=4 ? 16 : 0).
// B fragments use the same layout on transposed weights wT[n][k], so both
// sides are plain 4-byte LDS loads (coalesced to ds_load_b128).
// ---------------------------------------------------------------------------
__device__ __forceinline__ v16h frag_ld(const _Float16* s, int row, int ldk,
                                        int kk, int lane) {
  union { v16h v; unsigned u[8]; } f;
  const int hi = (lane & 16) ? 8 : 0;
  const unsigned* base = (const unsigned*)(s + row * ldk + kk);
#pragma unroll
  for (int r = 0; r < 8; ++r) {
    const int k0 = ((r & 3) << 1) + hi + ((r >> 2) << 4);
    f.u[r] = base[k0 >> 1];
  }
  return f.v;
}

// 16x16 tile for row-tile rt, K=128, with B fragments already in registers.
__device__ __forceinline__ v8f gemm_rows128(const _Float16* sA,
                                            const v16h bfr[4], int rt,
                                            int lane) {
  v8f acc = {};
  const int m = rt * 16 + (lane & 15);
#pragma unroll
  for (int kk = 0; kk < 4; ++kk) {
    v16h a = frag_ld(sA, m, HDIM, kk * 32, lane);
    acc = __builtin_amdgcn_wmma_f32_16x16x32_f16(
        false, a, false, bfr[kk], (short)0, acc, false, false);
  }
  return acc;
}

// ---------------------------------------------------------------------------
// Weight prep: f32 row-major W[K][N] -> f16 transposed wT[N][Kpad], zero pad.
// ---------------------------------------------------------------------------
__global__ __launch_bounds__(256) void wconv_kernel(const float* __restrict__ W,
                                                    _Float16* __restrict__ wT,
                                                    int K, int N, int Kpad) {
  int idx = blockIdx.x * 256 + threadIdx.x;
  if (idx >= N * Kpad) return;
  int nn = idx / Kpad, k = idx - nn * Kpad;
  wT[idx] = (k < K) ? (_Float16)W[k * N + nn] : (_Float16)0.f;
}

__global__ __launch_bounds__(256) void zero_kernel(float* __restrict__ p) {
  p[blockIdx.x * 256 + threadIdx.x] = 0.f;
}

// ---------------------------------------------------------------------------
// Node encoder: h = x @ W_enc + b_enc. 64 rows per block, B frags in regs.
// Row-tile validity (16-row granularity) is uniform -> scalar branch only.
// ---------------------------------------------------------------------------
__global__ __launch_bounds__(256) void node_enc_kernel(
    const float* __restrict__ x, const _Float16* __restrict__ wT,
    const float* __restrict__ bias, float* __restrict__ hout) {
  __shared__ _Float16 sA[64 * HDIM];
  __shared__ _Float16 sW[HDIM * HDIM];
  const int tid = threadIdx.x;
  const int row0 = blockIdx.x * 64;
  for (int i = tid; i < 64 * HDIM; i += 256) {
    const int row = row0 + (i >> 7);
    sA[i] = (row < NNODES) ? (_Float16)x[(size_t)row * HDIM + (i & 127)]
                           : (_Float16)0.f;
  }
  {
    const uint4* g = (const uint4*)wT;
    uint4* s4 = (uint4*)sW;
    for (int i = tid; i < (HDIM * HDIM) / 8; i += 256) s4[i] = g[i];
  }
  __syncthreads();
  const int wid = tid >> 5, lane = tid & 31;
  const int n = (lane & 15) + wid * 16;
  const int mb = (lane & 16) ? 8 : 0;
  v16h bfr[4];
#pragma unroll
  for (int kk = 0; kk < 4; ++kk) bfr[kk] = frag_ld(sW, n, HDIM, kk * 32, lane);
  const float bn = bias[n];
#pragma unroll
  for (int rt = 0; rt < 4; ++rt) {
    if (row0 + rt * 16 >= NNODES) break;  // uniform
    v8f acc = gemm_rows128(sA, bfr, rt, lane);
    float* op = hout + (size_t)(row0 + rt * 16 + mb) * HDIM + n;
#pragma unroll
    for (int r = 0; r < 8; ++r) op[(size_t)r * HDIM] = acc[r] + bn;
  }
}

// ---------------------------------------------------------------------------
// Edge encoder: ea = edge_attr @ W_ee + b_ee (f16 out). K=16 padded to 32,
// single WMMA per row-tile; 64 edges per block (E % 64 == 0, no guards).
// ---------------------------------------------------------------------------
__global__ __launch_bounds__(256) void edge_enc_kernel(
    const float* __restrict__ eattr, const _Float16* __restrict__ weeT,
    const float* __restrict__ bias, _Float16* __restrict__ ea) {
  __shared__ _Float16 sA[64 * 32];
  __shared__ _Float16 sW[HDIM * 32];
  const int tid = threadIdx.x;
  const int e0 = blockIdx.x * 64;
  for (int i = tid; i < 64 * 32; i += 256) {
    const int er = i >> 5, c = i & 31;
    sA[i] = (c < 16) ? (_Float16)eattr[(size_t)(e0 + er) * 16 + c]
                     : (_Float16)0.f;
  }
  {
    const uint4* g = (const uint4*)weeT;
    uint4* s4 = (uint4*)sW;
    for (int i = tid; i < (HDIM * 32) / 8; i += 256) s4[i] = g[i];
  }
  __syncthreads();
  const int wid = tid >> 5, lane = tid & 31;
  const int n = (lane & 15) + wid * 16;
  const int mb = (lane & 16) ? 8 : 0;
  const v16h b = frag_ld(sW, n, 32, 0, lane);
  const float bn = bias[n];
#pragma unroll
  for (int rt = 0; rt < 4; ++rt) {
    v16h a = frag_ld(sA, rt * 16 + (lane & 15), 32, 0, lane);
    v8f acc = {};
    acc = __builtin_amdgcn_wmma_f32_16x16x32_f16(false, a, false, b, (short)0,
                                                 acc, false, false);
    _Float16* op = ea + (size_t)(e0 + rt * 16 + mb) * HDIM + n;
#pragma unroll
    for (int r = 0; r < 8; ++r) op[(size_t)r * HDIM] = (_Float16)(acc[r] + bn);
  }
}

// ---------------------------------------------------------------------------
// Message + scatter: m = relu(h[src] + ea) * atten; atomicAdd into aggr[dst].
// One thread per (edge, 4 columns); exact grid, h/aggr resident in L2.
// ---------------------------------------------------------------------------
__global__ __launch_bounds__(256) void msg_kernel(
    const float* __restrict__ h, const _Float16* __restrict__ ea,
    const float* __restrict__ atten, const int* __restrict__ ei,
    float* __restrict__ aggr) {
  const int idx = blockIdx.x * 256 + threadIdx.x;  // < NEDGES*32 exactly
  const int e = idx >> 5;
  const int c = (idx & 31) << 2;
  const int src = ei[e];
  const int dst = ei[NEDGES + e];
  const float att = atten[e];
  const float4 hv = *(const float4*)(h + (size_t)src * HDIM + c);
  const v4h ev = *(const v4h*)(ea + (size_t)e * HDIM + c);
  float* ap = aggr + (size_t)dst * HDIM + c;
  atomicAdd(ap + 0, fmaxf(hv.x + (float)ev[0], 0.f) * att);
  atomicAdd(ap + 1, fmaxf(hv.y + (float)ev[1], 0.f) * att);
  atomicAdd(ap + 2, fmaxf(hv.z + (float)ev[2], 0.f) * att);
  atomicAdd(ap + 3, fmaxf(hv.w + (float)ev[3], 0.f) * att);
}

// ---------------------------------------------------------------------------
// Fused GINE layer MLP, 64 rows per block:
//   z = (1+eps)*h + aggr ; z@W1+b1 ; BN ; relu ; @W2+b2 ; relu -> hout
// Intermediate kept in LDS f16; W1/W2 B-fragments hoisted to registers.
// Row-tile validity is uniform (NNODES % 16 == 0) -> scalar branches only.
// ---------------------------------------------------------------------------
__global__ __launch_bounds__(256) void gine_layer_kernel(
    const float* __restrict__ hin, const float* __restrict__ aggr,
    const float* __restrict__ eps, const _Float16* __restrict__ w1T,
    const float* __restrict__ b1, const float* __restrict__ gamma,
    const float* __restrict__ beta, const float* __restrict__ mean,
    const float* __restrict__ var, const _Float16* __restrict__ w2T,
    const float* __restrict__ b2, float* __restrict__ hout) {
  __shared__ _Float16 sA[64 * HDIM];
  __shared__ _Float16 sZ[64 * HDIM];
  __shared__ _Float16 sW1[HDIM * HDIM];
  __shared__ _Float16 sW2[HDIM * HDIM];
  const int tid = threadIdx.x;
  const int row0 = blockIdx.x * 64;
  const float epsv = 1.0f + eps[0];
  for (int i = tid; i < 64 * HDIM; i += 256) {
    const int row = row0 + (i >> 7);
    if (row < NNODES) {
      const size_t gi = (size_t)row * HDIM + (i & 127);
      sA[i] = (_Float16)(epsv * hin[gi] + aggr[gi]);
    } else {
      sA[i] = (_Float16)0.f;
    }
  }
  {
    const uint4* g1 = (const uint4*)w1T;
    const uint4* g2 = (const uint4*)w2T;
    uint4* s1 = (uint4*)sW1;
    uint4* s2 = (uint4*)sW2;
    for (int i = tid; i < (HDIM * HDIM) / 8; i += 256) {
      s1[i] = g1[i];
      s2[i] = g2[i];
    }
  }
  __syncthreads();
  const int wid = tid >> 5, lane = tid & 31;
  const int n = (lane & 15) + wid * 16;
  const int mb = (lane & 16) ? 8 : 0;

  // GEMM1 + bias + BN + relu -> sZ (f16)
  v16h bfr[4];
#pragma unroll
  for (int kk = 0; kk < 4; ++kk) bfr[kk] = frag_ld(sW1, n, HDIM, kk * 32, lane);
  const float bb = b1[n], g = gamma[n], bt = beta[n], mu = mean[n];
  const float inv = rsqrtf(var[n] + 1e-5f);
#pragma unroll
  for (int rt = 0; rt < 4; ++rt) {
    if (row0 + rt * 16 >= NNODES) break;  // uniform
    v8f acc = gemm_rows128(sA, bfr, rt, lane);
    _Float16* zp = sZ + (rt * 16 + mb) * HDIM + n;
#pragma unroll
    for (int r = 0; r < 8; ++r) {
      float z = acc[r] + bb;
      z = (z - mu) * inv * g + bt;
      zp[r * HDIM] = (_Float16)fmaxf(z, 0.f);
    }
  }
  __syncthreads();

  // GEMM2 + bias + relu -> hout
#pragma unroll
  for (int kk = 0; kk < 4; ++kk) bfr[kk] = frag_ld(sW2, n, HDIM, kk * 32, lane);
  const float b2n = b2[n];
#pragma unroll
  for (int rt = 0; rt < 4; ++rt) {
    if (row0 + rt * 16 >= NNODES) break;  // uniform
    v8f acc = gemm_rows128(sZ, bfr, rt, lane);
    float* op = hout + (size_t)(row0 + rt * 16 + mb) * HDIM + n;
#pragma unroll
    for (int r = 0; r < 8; ++r) op[(size_t)r * HDIM] = fmaxf(acc[r] + b2n, 0.f);
  }
}

// ---------------------------------------------------------------------------
extern "C" void kernel_launch(void* const* d_in, const int* in_sizes, int n_in,
                              void* d_out, int out_size, void* d_ws,
                              size_t ws_size, hipStream_t stream) {
  const float* x     = (const float*)d_in[0];
  const int*   ei    = (const int*)d_in[1];
  // d_in[2] = batch (unused by the computation)
  const float* eattr = (const float*)d_in[3];
  const float* eatt  = (const float*)d_in[4];
  const float* Wenc  = (const float*)d_in[5];
  const float* benc  = (const float*)d_in[6];
  const float* Wee   = (const float*)d_in[7];
  const float* bee   = (const float*)d_in[8];
  const float* eps   = (const float*)d_in[9];
  const float* W1    = (const float*)d_in[10];
  const float* b1    = (const float*)d_in[11];
  const float* bng   = (const float*)d_in[12];
  const float* bnb   = (const float*)d_in[13];
  const float* bnm   = (const float*)d_in[14];
  const float* bnv   = (const float*)d_in[15];
  const float* W2    = (const float*)d_in[16];
  const float* b2    = (const float*)d_in[17];
  float* out = (float*)d_out;

  // workspace carve-up (256B aligned)
  char* p = (char*)d_ws;
  auto carve = [&](size_t bytes) {
    char* r = p;
    p += (bytes + 255) & ~(size_t)255;
    return r;
  };
  float*     hA    = (float*)carve((size_t)NNODES * HDIM * 4);
  float*     hB    = (float*)carve((size_t)NNODES * HDIM * 4);
  float*     aggr  = (float*)carve((size_t)NNODES * HDIM * 4);
  _Float16*  ea    = (_Float16*)carve((size_t)NEDGES * HDIM * 2);
  _Float16*  wencT = (_Float16*)carve((size_t)HDIM * HDIM * 2);
  _Float16*  weeT  = (_Float16*)carve((size_t)HDIM * 32 * 2);
  _Float16*  w1T   = (_Float16*)carve((size_t)NLAYERS * HDIM * HDIM * 2);
  _Float16*  w2T   = (_Float16*)carve((size_t)NLAYERS * HDIM * HDIM * 2);

  // 1) weight conversion (f32 row-major -> f16 transposed)
  wconv_kernel<<<(HDIM * HDIM + 255) / 256, 256, 0, stream>>>(Wenc, wencT, HDIM,
                                                              HDIM, HDIM);
  wconv_kernel<<<(HDIM * 32 + 255) / 256, 256, 0, stream>>>(Wee, weeT, 16, HDIM,
                                                            32);
  for (int i = 0; i < NLAYERS; ++i) {
    wconv_kernel<<<(HDIM * HDIM + 255) / 256, 256, 0, stream>>>(
        W1 + (size_t)i * HDIM * HDIM, w1T + (size_t)i * HDIM * HDIM, HDIM, HDIM,
        HDIM);
    wconv_kernel<<<(HDIM * HDIM + 255) / 256, 256, 0, stream>>>(
        W2 + (size_t)i * HDIM * HDIM, w2T + (size_t)i * HDIM * HDIM, HDIM, HDIM,
        HDIM);
  }

  // 2) encoders
  const int nblk = (NNODES + 63) / 64;
  node_enc_kernel<<<nblk, 256, 0, stream>>>(x, wencT, benc, hA);
  edge_enc_kernel<<<NEDGES / 64, 256, 0, stream>>>(eattr, weeT, bee, ea);

  // 3) layers
  float* hcur = hA;
  float* hnxt = hB;
  for (int i = 0; i < NLAYERS; ++i) {
    zero_kernel<<<(NNODES * HDIM) / 256, 256, 0, stream>>>(aggr);
    msg_kernel<<<(NEDGES * 32) / 256, 256, 0, stream>>>(hcur, ea, eatt, ei,
                                                        aggr);
    float* dst = (i == NLAYERS - 1) ? out : hnxt;
    gine_layer_kernel<<<nblk, 256, 0, stream>>>(
        hcur, aggr, eps + i, w1T + (size_t)i * HDIM * HDIM, b1 + i * HDIM,
        bng + i * HDIM, bnb + i * HDIM, bnm + i * HDIM, bnv + i * HDIM,
        w2T + (size_t)i * HDIM * HDIM, b2 + i * HDIM, dst);
    float* t = hcur;
    hcur = (dst == out) ? hcur : dst;
    hnxt = t;
  }
}
